// Transformer_32384053412064
// MI455X (gfx1250) — matrix operand order repaired and checked by reading, not compiled, not run on hardware
//
#include <hip/hip_runtime.h>
#include <hip/hip_bf16.h>

typedef __attribute__((ext_vector_type(16))) __bf16 v16bf;
typedef __attribute__((ext_vector_type(8)))  float  v8f;

#define N_HEADS 32
#define T_DIM   1024
#define S_DIM   1024
#define D_DIM   128
#define TT      16      // t-tile rows (WMMA M)
#define SB      32      // s-block (WMMA K for PV)
#define SP      34      // padded s pitch for LDS logits/probs tile (bf16)

// Fused dynamic-cross-head attention, two-pass flash style.
// grid = T/16 blocks, block = 1024 threads = 32 waves (wave m <-> head m).
__global__ __launch_bounds__(1024)
void dyn_attn_fused(const float* __restrict__ Q,  const float* __restrict__ Kt,
                    const float* __restrict__ Vt,
                    const float* __restrict__ sw_pre,  const float* __restrict__ qw1_pre,
                    const float* __restrict__ qw2_pre, const float* __restrict__ kw1_pre,
                    const float* __restrict__ kw2_pre, const float* __restrict__ qdd_pre,
                    const float* __restrict__ kdd_pre,
                    const float* __restrict__ sw_post,  const float* __restrict__ qw1_post,
                    const float* __restrict__ qw2_post, const float* __restrict__ kw1_post,
                    const float* __restrict__ kw2_post, const float* __restrict__ qdd_post,
                    const float* __restrict__ kdd_post,
                    float* __restrict__ Out)
{
    __shared__ __bf16 lds_p [N_HEADS][TT][SP];   // logits (pass1/2) then probs (pass2)
    __shared__ float  lds_qh[2][TT][SB];         // rank-2 query-side projections
    __shared__ float  lds_kh[2][TT][SB];         // rank-2 key-side projections
    __shared__ float  lds_swp[N_HEADS][N_HEADS]; // sw_pre
    __shared__ float  lds_swq[N_HEADS][N_HEADS]; // sw_post
    // k-side dynamic weights for the current s-block, staged via async-to-LDS DMA.
    // [0]=pre, [1]=post.  Read with half-wave-uniform addresses -> LDS broadcast.
    __shared__ __align__(16) float lds_kw2[2][SB][2 * N_HEADS];  // 16 KB
    __shared__ __align__(16) float lds_kdd[2][SB][N_HEADS];      //  8 KB

    const int tid   = threadIdx.x;
    const int head  = tid >> 5;        // wave id == head index
    const int lane  = tid & 31;
    const int lrow  = lane & 15;       // M (t-local) for A frags, N for B/C frags
    const int lhalf = lane >> 4;
    const int t0    = blockIdx.x * TT;
    const int tg    = t0 + lrow;       // this lane's global t row

    // stage 32x32 head-mix matrices once
    lds_swp[tid >> 5][tid & 31] = sw_pre [tid];
    lds_swq[tid >> 5][tid & 31] = sw_post[tid];

    // ---- preload Q A-fragments (M=t rows, K=d), f32 -> bf16 ----
    v16bf aq[4];
    #pragma unroll
    for (int c = 0; c < 4; ++c) {
        const float* qrow = Q + ((size_t)head * T_DIM + tg) * D_DIM + c * 32;
        #pragma unroll
        for (int j = 0; j < 16; ++j)
            aq[c][j] = (__bf16)qrow[j + 8 * (j >> 3) + 8 * lhalf];
    }

    // per-lane t-side dynamic params (t = tg, m = head fixed for this lane)
    const float qw2p0 = qw2_pre [(size_t)tg * 64 + head];
    const float qw2p1 = qw2_pre [(size_t)tg * 64 + 32 + head];
    const float qddp  = qdd_pre [(size_t)tg * 32 + head];
    const float qw2q0 = qw2_post[(size_t)tg * 64 + head];
    const float qw2q1 = qw2_post[(size_t)tg * 64 + 32 + head];
    const float qddq  = qdd_post[(size_t)tg * 32 + head];

    const int nsb = (t0 + TT + SB - 1) / SB;   // causal: only s-blocks touching s <= t0+15

    // ---- async-DMA the s-block's k-side weights into LDS (overlaps with QK^T) ----
    // kw2 block: 32 rows x 64 floats = 8KB contiguous;  kdd block: 32 x 32 = 4KB.
    auto stage_kside = [&](int s0, bool both) {
        if (head < 16) {                       // waves 0..15: pre weights
            const unsigned q = tid;            // 0..511, 16B granules
            unsigned long long src =
                (unsigned long long)(uintptr_t)((const char*)(kw2_pre + (size_t)s0 * 64) + q * 16);
            unsigned dst = (unsigned)(uintptr_t)((char*)&lds_kw2[0][0][0] + q * 16);
            asm volatile("global_load_async_to_lds_b128 %0, %1, off"
                         :: "v"(dst), "v"(src) : "memory");
            if (q < 256) {                     // waves 0..7: kdd pre
                unsigned long long s2 =
                    (unsigned long long)(uintptr_t)((const char*)(kdd_pre + (size_t)s0 * 32) + q * 16);
                unsigned d2 = (unsigned)(uintptr_t)((char*)&lds_kdd[0][0][0] + q * 16);
                asm volatile("global_load_async_to_lds_b128 %0, %1, off"
                             :: "v"(d2), "v"(s2) : "memory");
            }
        } else if (both) {                     // waves 16..31: post weights (pass 2 only)
            const unsigned q = tid - 512;      // 0..511
            unsigned long long src =
                (unsigned long long)(uintptr_t)((const char*)(kw2_post + (size_t)s0 * 64) + q * 16);
            unsigned dst = (unsigned)(uintptr_t)((char*)&lds_kw2[1][0][0] + q * 16);
            asm volatile("global_load_async_to_lds_b128 %0, %1, off"
                         :: "v"(dst), "v"(src) : "memory");
            if (q < 256) {                     // waves 16..23: kdd post
                unsigned long long s2 =
                    (unsigned long long)(uintptr_t)((const char*)(kdd_post + (size_t)s0 * 32) + q * 16);
                unsigned d2 = (unsigned)(uintptr_t)((char*)&lds_kdd[1][0][0] + q * 16);
                asm volatile("global_load_async_to_lds_b128 %0, %1, off"
                             :: "v"(d2), "v"(s2) : "memory");
            }
        }
    };

    // ---- QK^T for this wave's head into the shared tile (2 x 16-wide subtiles) ----
    auto logits_to_lds = [&](int s0) {
        #pragma unroll
        for (int sub = 0; sub < 2; ++sub) {
            const int sg = s0 + sub * 16 + lrow;            // B-frag: lane owns column s
            const float* krow = Kt + ((size_t)head * S_DIM + sg) * D_DIM;
            v8f c;
            #pragma unroll
            for (int r = 0; r < 8; ++r) c[r] = 0.f;
            #pragma unroll
            for (int ck = 0; ck < 4; ++ck) {                // K = d, 4 chunks of 32
                v16bf bk;
                #pragma unroll
                for (int j = 0; j < 16; ++j)
                    bk[j] = (__bf16)krow[ck * 32 + j + 8 * (j >> 3) + 8 * lhalf];
                c = __builtin_amdgcn_wmma_f32_16x16x32_bf16(false, aq[ck], false, bk,
                                                            (short)0, c, false, false);
            }
            __builtin_prefetch(krow + (size_t)SB * D_DIM, 0, 1);  // next s-block K row
            #pragma unroll
            for (int r = 0; r < 8; ++r)                     // C layout: M=r+8*lhalf, N=lrow
                lds_p[head][r + 8 * lhalf][sub * 16 + lrow] = (__bf16)c[r];
        }
    };

    // ---- per-point cross-head projection; outputs this lane's 16 A-frag elements ----
    auto cross_mix = [&](int s0, bool pre, float (&y)[16]) {
        asm volatile("s_wait_asynccnt 0x0" ::: "memory");    // k-side DMA landed
        __syncthreads();
        {   // cooperative rank-2 projections: 1024 threads x 512 points x {q,k}
            const int p   = tid & 511;
            const int grp = tid >> 9;
            const int pt  = p >> 5, ps = p & 31;
            const float* w1 = grp ? ((pre ? kw1_pre : kw1_post) + (size_t)(s0 + ps) * 64)
                                  : ((pre ? qw1_pre : qw1_post) + (size_t)(t0 + pt) * 64);
            float a0 = 0.f, a1 = 0.f;
            #pragma unroll 8
            for (int n = 0; n < N_HEADS; ++n) {
                const float x = (float)lds_p[n][pt][ps];
                a0 += x * w1[n];
                a1 += x * w1[32 + n];
            }
            if (grp == 0) { lds_qh[0][pt][ps] = a0; lds_qh[1][pt][ps] = a1; }
            else          { lds_kh[0][pt][ps] = a0; lds_kh[1][pt][ps] = a1; }
        }
        __syncthreads();
        const float (*swl)[N_HEADS] = pre ? lds_swp : lds_swq;
        const int    sel  = pre ? 0 : 1;
        const float  w20  = pre ? qw2p0 : qw2q0;
        const float  w21  = pre ? qw2p1 : qw2q1;
        const float  qdd  = pre ? qddp  : qddq;
        #pragma unroll
        for (int j = 0; j < 16; ++j) {
            const int sl = j + 8 * (j >> 3) + 8 * lhalf;    // A-frag element -> s-local
            const int sg = s0 + sl;
            const float pm = (float)lds_p[head][lrow][sl];
            float acc = pm;                                  // identity term
            #pragma unroll 8
            for (int n = 0; n < N_HEADS; ++n)                // 32x32 head mix
                acc += (float)lds_p[n][lrow][sl] * swl[n][head];
            acc += lds_qh[0][lrow][sl] * w20 + lds_qh[1][lrow][sl] * w21;
            acc += lds_kh[0][lrow][sl] * lds_kw2[sel][sl][head]
                 + lds_kh[1][lrow][sl] * lds_kw2[sel][sl][32 + head];
            acc += pm * (qdd + lds_kdd[sel][sl][head]);
            if (pre && sg > tg) acc = -1.0e30f;              // causal mask (additive min)
            y[j] = acc;
        }
    };

    // ================= pass 1: online softmax statistics =================
    float m_run = -3.0e38f, l_run = 0.f;                     // per lane = per row t
    for (int s0 = 0; s0 < nsb * SB; s0 += SB) {
        __syncthreads();                                      // protect LDS reuse
        stage_kside(s0, false);                               // overlaps with QK^T below
        logits_to_lds(s0);
        float y[16];
        cross_mix(s0, true, y);
        float tmax = -3.0e38f;
        #pragma unroll
        for (int j = 0; j < 16; ++j) tmax = fmaxf(tmax, y[j]);
        tmax = fmaxf(tmax, __shfl_xor(tmax, 16, 32));         // lanes l, l+16 share row t
        const float nm = fmaxf(m_run, tmax);
        float ls = 0.f;
        #pragma unroll
        for (int j = 0; j < 16; ++j) ls += __expf(y[j] - nm);
        ls += __shfl_xor(ls, 16, 32);
        l_run = l_run * __expf(m_run - nm) + ls;
        m_run = nm;
    }
    const float inv_l = (l_run > 0.f) ? 1.f / l_run : 0.f;

    // ================= pass 2: recompute, normalize, post-mix, PV =================
    v8f acc[8];
    #pragma unroll
    for (int dt = 0; dt < 8; ++dt)
        #pragma unroll
        for (int r = 0; r < 8; ++r) acc[dt][r] = 0.f;

    for (int s0 = 0; s0 < nsb * SB; s0 += SB) {
        __syncthreads();
        stage_kside(s0, true);
        logits_to_lds(s0);
        float y[16];
        cross_mix(s0, true, y);                               // pre-mixed masked logits
        __syncthreads();                                      // readers done before overwrite
        #pragma unroll
        for (int j = 0; j < 16; ++j) {                        // probs into shared tile
            const int sl = j + 8 * (j >> 3) + 8 * lhalf;
            lds_p[head][lrow][sl] = (__bf16)(__expf(y[j] - m_run) * inv_l);
        }
        float z[16];
        cross_mix(s0, false, z);                              // post projection on probs
        v16bf ap;
        #pragma unroll
        for (int j = 0; j < 16; ++j) ap[j] = (__bf16)z[j];
        #pragma unroll
        for (int dt = 0; dt < 8; ++dt) {                      // PV: A=probs(16x32) B=V(32x16)
            const float* vcol = Vt + ((size_t)head * S_DIM + s0 + 8 * lhalf) * D_DIM
                                   + dt * 16 + lrow;
            v16bf bv;
            #pragma unroll
            for (int j = 0; j < 16; ++j)
                bv[j] = (__bf16)vcol[(size_t)(j + 8 * (j >> 3)) * D_DIM];
            __builtin_prefetch(vcol + (size_t)SB * D_DIM, 0, 1);  // next s-block V
            acc[dt] = __builtin_amdgcn_wmma_f32_16x16x32_bf16(false, ap, false, bv,
                                                              (short)0, acc[dt], false, false);
        }
    }

    // ---- epilogue: C layout M=r+8*lhalf, N=lrow ----
    #pragma unroll
    for (int dt = 0; dt < 8; ++dt)
        #pragma unroll
        for (int r = 0; r < 8; ++r)
            Out[((size_t)head * T_DIM + t0 + r + 8 * lhalf) * D_DIM + dt * 16 + lrow] =
                acc[dt][r];
}

extern "C" void kernel_launch(void* const* d_in, const int* in_sizes, int n_in,
                              void* d_out, int out_size, void* d_ws, size_t ws_size,
                              hipStream_t stream) {
    (void)in_sizes; (void)n_in; (void)out_size; (void)d_ws; (void)ws_size;
    const float* Q        = (const float*)d_in[0];
    const float* K        = (const float*)d_in[1];
    const float* V        = (const float*)d_in[2];
    /* d_in[3] = atten_mask: causal, applied analytically inside the kernel */
    const float* sw_pre   = (const float*)d_in[4];
    const float* qw1_pre  = (const float*)d_in[5];
    const float* qw2_pre  = (const float*)d_in[6];
    const float* kw1_pre  = (const float*)d_in[7];
    const float* kw2_pre  = (const float*)d_in[8];
    const float* qdd_pre  = (const float*)d_in[9];
    const float* kdd_pre  = (const float*)d_in[10];
    const float* sw_post  = (const float*)d_in[11];
    const float* qw1_post = (const float*)d_in[12];
    const float* qw2_post = (const float*)d_in[13];
    const float* kw1_post = (const float*)d_in[14];
    const float* kw2_post = (const float*)d_in[15];
    const float* qdd_post = (const float*)d_in[16];
    const float* kdd_post = (const float*)d_in[17];
    float* out = (float*)d_out;

    dim3 grid(T_DIM / TT);   // 64 t-tiles
    dim3 block(1024);        // 32 waves = 32 heads
    hipLaunchKernelGGL(dyn_attn_fused, grid, block, 0, stream,
                       Q, K, V,
                       sw_pre, qw1_pre, qw2_pre, kw1_pre, kw2_pre, qdd_pre, kdd_pre,
                       sw_post, qw1_post, qw2_post, kw1_post, kw2_post, qdd_post, kdd_post,
                       out);
}
